// SelfAttention_52261162058620
// MI455X (gfx1250) — compile-verified, matrix operand order checked by reading, not run on hardware
//
#include <hip/hip_runtime.h>
#include <hip/hip_bf16.h>
#include <math.h>
#include <stdint.h>

#define USE_TDM 1

typedef __attribute__((ext_vector_type(16))) _Float16 v16h;
typedef __attribute__((ext_vector_type(8)))  _Float16 v8h;
typedef __attribute__((ext_vector_type(8)))  float    v8f;

#define NH 32
#define HD 128
#define NG 2
#define REP (NH / NG)
#define B_  2
#define SQ  2048
#define HIDDEN 4096
#define QKV_OUT (NH * HD + 2 * NG * HD)   // 4608
#define ROT_DIM 64

// Build a 16x32 WMMA operand from a LINEAR 32-half K-chunk in LDS.
// ISA layout: lane-half 0 wants K {0..7, 16..23}; half 1 wants {8..15, 24..31}.
// Both are two contiguous 16B groups -> two ds_load_b128.
__device__ __forceinline__ v16h ld_op(const _Float16* rowbase, int hid) {
  const v8h lo = *(const v8h*)(rowbase + hid * 8);        // K hid*8 .. +8
  const v8h hi = *(const v8h*)(rowbase + 16 + hid * 8);   // K 16+hid*8 .. +8
  v16h r;
#pragma unroll
  for (int u = 0; u < 8; ++u) { r[u] = lo[u]; r[u + 8] = hi[u]; }
  return r;
}

#if USE_TDM
typedef __attribute__((ext_vector_type(4))) unsigned int u32x4;
typedef __attribute__((ext_vector_type(8))) unsigned int u32x8;

// Issue a 2-D TENSOR_LOAD_TO_LDS (f16 elements). One call per wave (wave 0).
// D# group0: count=1 | lds_addr | global_addr(57b) | type=2 ("image").
// D# group1: data_size=2B, tensor_dim0/1, tile_dim0/1, tensor_dim0_stride.
__device__ __forceinline__ void tdm_load_2d(void* lds_dst, const void* gsrc,
                                            unsigned tdim0, unsigned tdim1,
                                            unsigned long long stride0,
                                            unsigned tile0, unsigned tile1)
{
  const unsigned long long ga = (unsigned long long)(uintptr_t)gsrc;
  const unsigned lds = (unsigned)(uintptr_t)lds_dst;   // low 32b = LDS offset
  u32x4 g0;
  g0[0] = 1u;                                   // count=1 (valid user D#)
  g0[1] = lds;                                  // lds_addr (bytes)
  g0[2] = (unsigned)ga;                         // global_addr[31:0]
  g0[3] = ((unsigned)(ga >> 32) & 0x01FFFFFFu) | (2u << 30); // [56:32]|type=2
  u32x8 g1;
  g1[0] = 1u << 16;                             // data_size=1 (2 bytes)
  g1[1] = (tdim0 & 0xFFFFu) << 16;              // tensor_dim0[15:0] @ bit48
  g1[2] = (tdim0 >> 16) | ((tdim1 & 0xFFFFu) << 16);
  g1[3] = (tdim1 >> 16) | (tile0 << 16);        // tile_dim0 @ bit112
  g1[4] = tile1;                                // tile_dim1 (tile_dim2 = 0)
  g1[5] = (unsigned)stride0;                    // tensor_dim0_stride[31:0]
  g1[6] = (unsigned)(stride0 >> 32) & 0xFFFFu;  // stride[47:32]; dim1_stride=0
  g1[7] = 0u;
  asm volatile("tensor_load_to_lds %0, %1" :: "s"(g0), "s"(g1) : "memory");
}
#endif

// -------------------------------------------------------------------------
// f32 -> f16 bulk convert (hidden states).
// -------------------------------------------------------------------------
__global__ __launch_bounds__(256) void f32_to_f16_kernel(
    const float* __restrict__ X, _Float16* __restrict__ Y, long long n)
{
  const long long i = ((long long)blockIdx.x * blockDim.x + threadIdx.x) * 8;
  if (i + 8 > n) return;
  v8h g;
#pragma unroll
  for (int u = 0; u < 8; ++u) g[u] = (_Float16)X[i + u];
  *(v8h*)&Y[i] = g;
}

// -------------------------------------------------------------------------
// W[K][N] f32  ->  Wt[N][K] f16, 64x64 LDS tile transpose (coalesced both ways)
// -------------------------------------------------------------------------
__global__ __launch_bounds__(256) void transpose_f32_to_f16_kernel(
    const float* __restrict__ W, _Float16* __restrict__ Wt, int K, int N)
{
  __shared__ _Float16 T[64][65];
  const int k0 = blockIdx.x * 64;
  const int n0 = blockIdx.y * 64;
  for (int idx = threadIdx.x; idx < 64 * 64; idx += 256) {
    const int r = idx >> 6, c = idx & 63;
    T[r][c] = (_Float16)W[(size_t)(k0 + r) * N + n0 + c];
  }
  __syncthreads();
  for (int idx = threadIdx.x; idx < 64 * 64; idx += 256) {
    const int r = idx >> 6, c = idx & 63;      // r: n-local, c: k-local
    Wt[(size_t)(n0 + r) * K + k0 + c] = T[c][r];
  }
}

// -------------------------------------------------------------------------
// GEMM: C[M,N] = A[M,K] * Bt[N,K]^T (+bias). All-f16 operands, f32 accum.
// 256 thr = 8 waves, 128x128 tile, BK=32, double-buffered LDS staged by TDM
// (wave 0 issues tensor_load_to_lds for the next tile while all 8 waves run
// 8 WMMAs on the current one; s_wait_tensorcnt + barrier hand over).
// -------------------------------------------------------------------------
template <typename TC>
__global__ __launch_bounds__(256) void gemm_f16_wmma_kernel(
    const _Float16* __restrict__ A, const _Float16* __restrict__ Bt,
    const float* __restrict__ bias, TC* __restrict__ C,
    int M, int N, int K)
{
  __shared__ alignas(16) _Float16 As[2][128][32];
  __shared__ alignas(16) _Float16 Bs[2][128][32];

  const int tid  = threadIdx.x;
  const int lane = tid & 31;
  const int wave = tid >> 5;
  const int i    = lane & 15;
  const int hid  = lane >> 4;
  const int row0 = blockIdx.x * 128;
  const int col0 = blockIdx.y * 128;

  v8f acc[8] = {};
  const int nsteps = K >> 5;

#if USE_TDM
  if (wave == 0) {
    tdm_load_2d(&As[0][0][0], A  + (size_t)row0 * K, K, M - row0, K, 32, 128);
    tdm_load_2d(&Bs[0][0][0], Bt + (size_t)col0 * K, K, N - col0, K, 32, 128);
    __builtin_amdgcn_s_wait_tensorcnt(0);
  }
  __syncthreads();
#else
  {
    const int r = tid >> 1, s2 = tid & 1;
    const _Float16* pa = A  + (size_t)(row0 + r) * K + s2 * 16;
    const _Float16* pb = Bt + (size_t)(col0 + r) * K + s2 * 16;
    *(v8h*)&As[0][r][s2 * 16]     = *(const v8h*)pa;
    *(v8h*)&As[0][r][s2 * 16 + 8] = *(const v8h*)(pa + 8);
    *(v8h*)&Bs[0][r][s2 * 16]     = *(const v8h*)pb;
    *(v8h*)&Bs[0][r][s2 * 16 + 8] = *(const v8h*)(pb + 8);
  }
  __syncthreads();
#endif

  for (int s = 0; s < nsteps; ++s) {
    const int par = s & 1;
#if USE_TDM
    if (s + 1 < nsteps && wave == 0) {   // async-prefetch next tile via TDM
      const int kn = (s + 1) * 32;
      tdm_load_2d(&As[1 - par][0][0], A  + (size_t)row0 * K + kn,
                  K, M - row0, K, 32, 128);
      tdm_load_2d(&Bs[1 - par][0][0], Bt + (size_t)col0 * K + kn,
                  K, N - col0, K, 32, 128);
    }
#else
    if (s + 1 < nsteps) {
      const int kn = (s + 1) * 32;
      const int r = tid >> 1, s2 = tid & 1;
      const _Float16* pa = A  + (size_t)(row0 + r) * K + kn + s2 * 16;
      const _Float16* pb = Bt + (size_t)(col0 + r) * K + kn + s2 * 16;
      *(v8h*)&As[1 - par][r][s2 * 16]     = *(const v8h*)pa;
      *(v8h*)&As[1 - par][r][s2 * 16 + 8] = *(const v8h*)(pa + 8);
      *(v8h*)&Bs[1 - par][r][s2 * 16]     = *(const v8h*)pb;
      *(v8h*)&Bs[1 - par][r][s2 * 16 + 8] = *(const v8h*)(pb + 8);
    }
#endif
    const v16h a = ld_op(&As[par][wave * 16 + i][0], hid);
#pragma unroll
    for (int ct = 0; ct < 8; ++ct) {
      const v16h b = ld_op(&Bs[par][ct * 16 + i][0], hid);
      acc[ct] = __builtin_amdgcn_wmma_f32_16x16x32_f16(false, a, false, b,
                                                       (short)0, acc[ct],
                                                       false, false);
    }
    __syncthreads();                     // all reads of buf[par] done
#if USE_TDM
    if (s + 1 < nsteps && wave == 0) __builtin_amdgcn_s_wait_tensorcnt(0);
    __syncthreads();                     // next tile visible to all waves
#endif
  }

#pragma unroll
  for (int ct = 0; ct < 8; ++ct) {
#pragma unroll
    for (int r = 0; r < 8; ++r) {
      const int row = row0 + wave * 16 + r + 8 * hid;
      const int col = col0 + ct * 16 + i;
      float v = acc[ct][r];
      if (bias) v += bias[col];
      C[(size_t)row * N + col] = (TC)v;
    }
  }
}

// -------------------------------------------------------------------------
// RoPE + split. Q[b,NH,sq,hd], K[b,NG,sq,hd] row-major; V written TRANSPOSED
// as Vt[b,NG,hd,sq] so the attention V tile is a dense 2-D tile.
// -------------------------------------------------------------------------
__global__ __launch_bounds__(256) void rope_split_kernel(
    const _Float16* __restrict__ mixed, const float* __restrict__ rope,
    _Float16* __restrict__ Q, _Float16* __restrict__ Kh, _Float16* __restrict__ Vt)
{
  const long long idx   = (long long)blockIdx.x * blockDim.x + threadIdx.x;
  const long long total = (long long)B_ * SQ * (NH + 2 * NG) * (HD / 2);
  if (idx >= total) return;
  const int pair = (int)(idx & (HD / 2 - 1));
  long long t = idx >> 6;
  const int slot = (int)(t % (NH + 2 * NG));    // 0..31 Q, 32..33 K, 34..35 V
  t /= (NH + 2 * NG);
  const int s  = (int)(t % SQ);
  const int bb = (int)(t / SQ);
  const int d0 = pair * 2;

  const size_t src = ((size_t)(bb * SQ + s)) * QKV_OUT + (size_t)slot * HD + d0;
  const float x0 = (float)mixed[src], x1 = (float)mixed[src + 1];
  float y0 = x0, y1 = x1;
  if (slot < NH + NG && d0 < ROT_DIM) {
    const float c0 = rope[((size_t)s * (ROT_DIM / 2) + pair) * 2 + 0];
    const float c1 = rope[((size_t)s * (ROT_DIM / 2) + pair) * 2 + 1];
    y0 = x0 * c0 - x1 * c1;
    y1 = x1 * c0 + x0 * c1;
  }
  if (slot < NH) {
    _Float16* dst = Q + (((size_t)(bb * NH + slot) * SQ + s) * HD + d0);
    dst[0] = (_Float16)y0; dst[1] = (_Float16)y1;
  } else if (slot < NH + NG) {
    _Float16* dst = Kh + (((size_t)(bb * NG + (slot - NH)) * SQ + s) * HD + d0);
    dst[0] = (_Float16)y0; dst[1] = (_Float16)y1;
  } else {
    const size_t vb = (size_t)(bb * NG + (slot - NH - NG)) * HD;
    Vt[(vb + d0)     * SQ + s] = (_Float16)y0;
    Vt[(vb + d0 + 1) * SQ + s] = (_Float16)y1;
  }
}

// -------------------------------------------------------------------------
// Causal flash attention with GQA. Block = 8 waves, 16-query tile, 32-key
// step. Q/K/V tiles staged by TDM (wave 0) into linear LDS; WMMA operands
// read as dual ds_load_b128.
// -------------------------------------------------------------------------
__global__ __launch_bounds__(256) void flash_attn_kernel(
    const _Float16* __restrict__ Q, const _Float16* __restrict__ Kh,
    const _Float16* __restrict__ Vt, _Float16* __restrict__ Ctx)
{
  __shared__ alignas(16) _Float16 Qs[16][HD];
  __shared__ alignas(16) _Float16 Ks[32][HD];
  __shared__ alignas(16) _Float16 Vst[HD][32];   // [hd][key] (from Vt)
  __shared__ float    Sf[16][32];
  __shared__ alignas(16) _Float16 Ps[16][32];
  __shared__ float    row_m[16], row_l[16], row_scale[16];

  const int tid  = threadIdx.x;
  const int lane = tid & 31;
  const int wave = tid >> 5;
  const int i    = lane & 15;
  const int hid  = lane >> 4;

  const int q0 = blockIdx.x * 16;
  const int h  = blockIdx.y;
  const int bb = blockIdx.z;
  const int g  = h / REP;

  const _Float16* Qg  = Q  + ((size_t)(bb * NH + h) * SQ) * HD;
  const _Float16* Kg  = Kh + ((size_t)(bb * NG + g) * SQ) * HD;
  const _Float16* Vtg = Vt + ((size_t)(bb * NG + g) * HD) * SQ;

#if USE_TDM
  if (wave == 0)   // completion covered by the first in-loop tensorcnt wait
    tdm_load_2d(&Qs[0][0], Qg + (size_t)q0 * HD, HD, SQ - q0, HD, HD, 16);
#else
  if (tid < 128) {
    const int r = tid >> 3, seg = tid & 7;
    const _Float16* p = Qg + (size_t)(q0 + r) * HD + seg * 16;
    *(v8h*)&Qs[r][seg * 16]     = *(const v8h*)p;
    *(v8h*)&Qs[r][seg * 16 + 8] = *(const v8h*)(p + 8);
  }
#endif
  if (tid < 16) { row_m[tid] = -1e30f; row_l[tid] = 0.0f; }

  v8f acc_o = {};
  const float scale = 1.0f / sqrtf((float)HD);   // coeff == 1 for layer 1

  for (int k0 = 0; k0 <= q0 + 15; k0 += 32) {
    __syncthreads();   // previous iteration's Ks/Vst/Ps fully consumed
#if USE_TDM
    if (wave == 0) {
      tdm_load_2d(&Ks[0][0],  Kg + (size_t)k0 * HD, HD, SQ - k0, HD, HD, 32);
      tdm_load_2d(&Vst[0][0], Vtg + k0,             SQ, HD,      SQ, 32, HD);
      __builtin_amdgcn_s_wait_tensorcnt(0);
    }
#else
    {
      const int r = tid >> 3, seg = tid & 7;
      const _Float16* p = Kg + (size_t)(k0 + r) * HD + seg * 16;
      *(v8h*)&Ks[r][seg * 16]     = *(const v8h*)p;
      *(v8h*)&Ks[r][seg * 16 + 8] = *(const v8h*)(p + 8);
      const int n = tid >> 1, s2 = tid & 1;
      const _Float16* pv = Vtg + (size_t)n * SQ + k0 + s2 * 16;
      *(v8h*)&Vst[n][s2 * 16]     = *(const v8h*)pv;
      *(v8h*)&Vst[n][s2 * 16 + 8] = *(const v8h*)(pv + 8);
    }
#endif
    __syncthreads();

    if (wave < 2) {                // S = Q * K^T for keys [k0+16*wave, +16)
      v8f s = {};
#pragma unroll
      for (int kk = 0; kk < HD; kk += 32) {
        const v16h a = ld_op(&Qs[i][kk], hid);
        const v16h b = ld_op(&Ks[wave * 16 + i][kk], hid);
        s = __builtin_amdgcn_wmma_f32_16x16x32_f16(false, a, false, b,
                                                   (short)0, s, false, false);
      }
#pragma unroll
      for (int r = 0; r < 8; ++r) {
        const int rr = r + 8 * hid;                 // query row in tile
        const int kq = k0 + wave * 16 + i;          // global key index
        float v = s[r] * scale;
        if (kq > q0 + rr) v = -1e30f;               // causal mask
        Sf[rr][wave * 16 + i] = v;
      }
    }
    __syncthreads();

    if (tid < 16) {                // online softmax, one row per lane
      const int row = tid;
      const float m_old = row_m[row];
      float mx = m_old;
      for (int j = 0; j < 32; ++j) mx = fmaxf(mx, Sf[row][j]);
      const float corr = __expf(m_old - mx);
      float l = row_l[row] * corr;
      for (int j = 0; j < 32; ++j) {
        const float p = __expf(Sf[row][j] - mx);
        Ps[row][j] = (_Float16)p;
        l += p;
      }
      row_m[row] = mx; row_l[row] = l; row_scale[row] = corr;
    }
    __syncthreads();

    {                              // O += P * V for this wave's 16 columns
      const v16h a = ld_op(&Ps[i][0], hid);
      const v16h b = ld_op(&Vst[wave * 16 + i][0], hid);
#pragma unroll
      for (int r = 0; r < 8; ++r) acc_o[r] *= row_scale[r + 8 * hid];
      acc_o = __builtin_amdgcn_wmma_f32_16x16x32_f16(false, a, false, b,
                                                     (short)0, acc_o,
                                                     false, false);
    }
  }

  __syncthreads();
#pragma unroll
  for (int r = 0; r < 8; ++r) {
    const int row = r + 8 * hid;
    const float o = acc_o[r] / row_l[row];
    Ctx[((size_t)(bb * SQ + q0 + row)) * (NH * HD) + (size_t)h * HD + wave * 16 + i]
        = (_Float16)o;
  }
}

extern "C" void kernel_launch(void* const* d_in, const int* in_sizes, int n_in,
                              void* d_out, int out_size, void* d_ws, size_t ws_size,
                              hipStream_t stream) {
  const float* hidden = (const float*)d_in[0];
  const float* rope   = (const float*)d_in[1];
  const float* Wqkv   = (const float*)d_in[2];
  const float* bqkv   = (const float*)d_in[3];
  const float* Wdense = (const float*)d_in[4];
  float* out = (float*)d_out;
  (void)in_sizes; (void)n_in; (void)out_size; (void)ws_size;

  const size_t M = (size_t)B_ * SQ;   // 4096 token rows
  char* ws = (char*)d_ws;
  _Float16* Ah   = (_Float16*)ws;  ws += M * HIDDEN * sizeof(_Float16);                 // 32 MB
  _Float16* Wq_t = (_Float16*)ws;  ws += (size_t)QKV_OUT * HIDDEN * sizeof(_Float16);   // 36 MB
  _Float16* Wd_t = (_Float16*)ws;  ws += (size_t)HIDDEN * HIDDEN * sizeof(_Float16);    // 32 MB
  _Float16* mixh = (_Float16*)ws;  ws += M * QKV_OUT * sizeof(_Float16);                // 37.7 MB
  _Float16* Qh   = (_Float16*)ws;  ws += (size_t)B_ * NH * SQ * HD * sizeof(_Float16);  // 32 MB
  _Float16* Kk   = (_Float16*)ws;  ws += (size_t)B_ * NG * SQ * HD * sizeof(_Float16);  //  2 MB
  _Float16* Vt   = (_Float16*)ws;  ws += (size_t)B_ * NG * SQ * HD * sizeof(_Float16);  //  2 MB
  _Float16* ctx  = (_Float16*)ws;  ws += M * (size_t)(NH * HD) * sizeof(_Float16);      // 32 MB

  // 0) precision/layout prep: hidden -> f16, weights -> f16 transposed [N][K]
  f32_to_f16_kernel<<<(unsigned)((M * HIDDEN / 8 + 255) / 256), 256, 0, stream>>>(
      hidden, Ah, (long long)(M * HIDDEN));
  transpose_f32_to_f16_kernel<<<dim3(HIDDEN / 64, QKV_OUT / 64), 256, 0, stream>>>(
      Wqkv, Wq_t, HIDDEN, QKV_OUT);
  transpose_f32_to_f16_kernel<<<dim3(HIDDEN / 64, HIDDEN / 64), 256, 0, stream>>>(
      Wdense, Wd_t, HIDDEN, HIDDEN);

  // 1) QKV projection (+bias): mixh[M, 4608] f16
  gemm_f16_wmma_kernel<_Float16>
      <<<dim3((unsigned)(M / 128), QKV_OUT / 128), 256, 0, stream>>>(
          Ah, Wq_t, bqkv, mixh, (int)M, QKV_OUT, HIDDEN);

  // 2) RoPE + head split (V transposed)
  {
    const long long total = (long long)B_ * SQ * (NH + 2 * NG) * (HD / 2);
    rope_split_kernel<<<(unsigned)((total + 255) / 256), 256, 0, stream>>>(
        mixh, rope, Qh, Kk, Vt);
  }

  // 3) causal flash attention -> ctx[M, 4096] f16
  flash_attn_kernel<<<dim3(SQ / 16, NH, B_), 256, 0, stream>>>(Qh, Kk, Vt, ctx);

  // 4) output projection -> d_out[M, 4096] f32
  gemm_f16_wmma_kernel<float>
      <<<dim3((unsigned)(M / 128), HIDDEN / 128), 256, 0, stream>>>(
          ctx, Wd_t, nullptr, out, (int)M, HIDDEN, HIDDEN);
}
